// ScaledDotProductAttention_79834852098494
// MI455X (gfx1250) — matrix operand order repaired and checked by reading, not compiled, not run on hardware
//
#include <hip/hip_runtime.h>

// ---------------------------------------------------------------------------
// ScaledDotProductAttention (single head), B=8 T=2048 C=1024 H=64,
// scale = 1/sqrt(2048)  (module scales by block_size, per reference).
//
// bf16 WMMA + fp32 accumulation, flash-style online softmax, transposed
// attention orientation (S^T = K Q^T, O^T = V^T P^T) so softmax stats are
// per-lane scalars.  Projection kernel stages its weight tiles in LDS with
// GLOBAL_LOAD_ASYNC_TO_LDS_B128 (ASYNCcnt) shared by the whole block.
// ---------------------------------------------------------------------------

#define Bn 8
#define Tn 2048
#define Cn 1024
#define Hn 64
#define SCALE 0.022097086912079612f   // 2048^-0.5
#define NEGINF (-1.0e30f)

typedef __bf16 bf16_t;
typedef __attribute__((ext_vector_type(16))) __bf16 v16bf;
typedef __attribute__((ext_vector_type(8)))  __bf16 v8bf;
typedef __attribute__((ext_vector_type(8)))  float  v8f;

static __device__ inline v8f wmma_bf16(v16bf a, v16bf b, v8f c) {
  // (neg_a, A, neg_b, B, c_mod, C, reuse_a, reuse_b)
  return __builtin_amdgcn_wmma_f32_16x16x32_bf16(false, a, false, b,
                                                 (short)0, c, false, false);
}

static __device__ inline v16bf join8(v8bf lo, v8bf hi) {
  return __builtin_shufflevector(lo, hi, 0,1,2,3,4,5,6,7,8,9,10,11,12,13,14,15);
}

// A-fragment (16xK): p already includes +16*(lane>>4); 16 contiguous elems.
static __device__ inline v16bf load_afrag(const bf16_t* p) {
  v8bf lo = *(const v8bf*)(p);
  v8bf hi = *(const v8bf*)(p + 8);
  return join8(lo, hi);
}

// B-fragment (Kx16): p already includes +16*(lane>>4); 16 contiguous elems.
static __device__ inline v16bf load_bfrag(const bf16_t* p) {
  v8bf lo = *(const v8bf*)(p);
  v8bf hi = *(const v8bf*)(p + 8);
  return join8(lo, hi);
}

// Async copy of 16 bytes global -> LDS, tracked by ASYNCcnt.
// LDS destination = byte offset (low 32 bits of a generic LDS pointer: the
// flat LDS aperture keeps the offset in addr[31:0]).
static __device__ inline void async_copy_b128(const void* gptr, void* lptr) {
  unsigned ldsoff = (unsigned)(size_t)lptr;
  asm volatile("global_load_async_to_lds_b128 %0, %1, off"
               :: "v"(ldsoff), "v"(gptr) : "memory");
}

// ---------------------------------------------------------------------------
// Kernel 0: convert the three 64x1024 f32 weight matrices to bf16 in ws.
// ---------------------------------------------------------------------------
__global__ __launch_bounds__(256) void cvt_weights_kernel(
    const float* __restrict__ Wq, const float* __restrict__ Wk,
    const float* __restrict__ Wv, bf16_t* __restrict__ out) {
  int i = blockIdx.x * 256 + threadIdx.x;           // 0 .. 65535
  const int N = Hn * Cn;                            // 65536
  if (i < N) {
    out[i]         = (__bf16)Wq[i];
    out[N + i]     = (__bf16)Wk[i];
    out[2 * N + i] = (__bf16)Wv[i];
  }
}

// ---------------------------------------------------------------------------
// Kernel 1: QKV projection.  4 waves/block, one 16-row tile each.
// Per 32-wide K-step: the block cooperatively async-copies the 12 KB of
// Wq/Wk/Wv tiles into LDS (6x b128 per thread, ASYNCcnt), then each wave
// reads B-fragments from LDS (ds_load_b128) and fires 12 WMMAs.
// Q,K stored row-major bf16 [B*T,64]; V stored transposed Vt[b][h][t].
// ---------------------------------------------------------------------------
__global__ __launch_bounds__(128) void qkv_proj_kernel(
    const float* __restrict__ x,
    const bf16_t* __restrict__ Wq, const bf16_t* __restrict__ Wk,
    const bf16_t* __restrict__ Wv,
    bf16_t* __restrict__ Q, bf16_t* __restrict__ K, bf16_t* __restrict__ Vt) {
  // LDS weight tile: [mat][h=64][c=32] bf16 = 12 KB.
  __shared__ bf16_t wlds[3 * 64 * 32];

  const int tid   = threadIdx.x;
  const int lane  = tid & 31;
  const int wave  = tid >> 5;
  const int tile  = blockIdx.x * 4 + wave;          // 0 .. B*T/16 - 1
  const long row0 = (long)tile * 16;                // flat row in [0, B*T)
  const int l15   = lane & 15;
  const int khalf = lane >> 4;
  const int koffA = khalf * 16;                     // A-frag K offset
  const int koffB = khalf * 16;                     // B-frag K offset
  const int rbase = khalf * 8;                      // C-layout row base

  const bf16_t* Wmat[3] = {Wq, Wk, Wv};

  v8f accQ[4], accK[4], accV[4];
  const v8f z = {0.f,0.f,0.f,0.f,0.f,0.f,0.f,0.f};
  #pragma unroll
  for (int nt = 0; nt < 4; ++nt) { accQ[nt] = z; accK[nt] = z; accV[nt] = z; }

  for (int kc = 0; kc < Cn; kc += 32) {
    // ---- block-cooperative async stage of the 3 weight tiles ------------
    // chunk = i*128 + tid in [0,768): mat = chunk/256 (= i>>1, static),
    // rem = chunk%256 -> h = rem/4, 16B-quarter q = rem%4.
    #pragma unroll
    for (int i = 0; i < 6; ++i) {
      const int mat = i >> 1;                       // compile-time
      const int rem = ((i & 1) << 7) | tid;         // 0..255
      const int h   = rem >> 2;
      const int q   = rem & 3;
      async_copy_b128(Wmat[mat] + (long)h * Cn + kc + q * 8,
                      &wlds[(mat << 11) + h * 32 + q * 8]);
    }

    // ---- A fragment: x rows, f32 -> bf16 (overlaps the async copies) ----
    const float* xr = x + (row0 + l15) * Cn + kc + koffA;
    __builtin_prefetch(xr + 32, 0, 1);              // next K-chunk
    v16bf a;
    #pragma unroll
    for (int e = 0; e < 8; ++e) a[e]     = (__bf16)xr[e];
    #pragma unroll
    for (int e = 0; e < 8; ++e) a[8 + e] = (__bf16)xr[8 + e];

    asm volatile("s_wait_asynccnt 0" ::: "memory"); // own copies landed
    __syncthreads();                                // everyone's copies landed

    // ---- B-fragments from LDS + 12 WMMAs --------------------------------
    #pragma unroll
    for (int nt = 0; nt < 4; ++nt) {
      const int hrow = (l15 + 16 * nt) * 32 + koffB;
      accQ[nt] = wmma_bf16(a, load_bfrag(&wlds[          hrow]), accQ[nt]);
      accK[nt] = wmma_bf16(a, load_bfrag(&wlds[ (1<<11) + hrow]), accK[nt]);
      accV[nt] = wmma_bf16(a, load_bfrag(&wlds[ (2<<11) + hrow]), accV[nt]);
    }
    __syncthreads();                                // reads done before overwrite
  }

  const int  bidx = (int)(row0 / Tn);
  const int  tloc = (int)(row0 % Tn);
  bf16_t* vtb = Vt + (long)bidx * Hn * Tn;
  #pragma unroll
  for (int nt = 0; nt < 4; ++nt) {
    #pragma unroll
    for (int r = 0; r < 8; ++r) {
      const long idx = (row0 + r + rbase) * Hn + l15 + 16 * nt;
      Q[idx] = (__bf16)accQ[nt][r];
      K[idx] = (__bf16)accK[nt][r];
      // V transposed: Vt[b][h][t]
      vtb[(long)(l15 + 16 * nt) * Tn + tloc + r + rbase] = (__bf16)accV[nt][r];
    }
  }
}

// ---------------------------------------------------------------------------
// Kernel 2: causal flash attention, TRANSPOSED orientation.
// One wave per 16-query tile.  S^T = K Q^T : keys in registers, queries in
// lanes -> per-lane softmax stats (single xor-16 shuffle merges key halves).
// P staged [query][key] in LDS with vector b128 ds ops; O^T = V^T P^T.
// ---------------------------------------------------------------------------
__global__ __launch_bounds__(32) void flash_attn_kernel(
    const bf16_t* __restrict__ Q, const bf16_t* __restrict__ K,
    const bf16_t* __restrict__ Vt, float* __restrict__ out) {
  __shared__ __bf16 pbuf[16 * 32];                  // P: [query][key] (1 KB)

  const int lane  = threadIdx.x & 31;
  const int l15   = lane & 15;
  const int khalf = lane >> 4;
  const int koffA = khalf * 16;
  const int koffB = khalf * 16;
  const int rbase = khalf * 8;                      // C-layout row base

  const int tile = blockIdx.x;                      // 0 .. B*T/16 - 1
  const int b    = tile / (Tn / 16);
  const int t0   = (tile % (Tn / 16)) * 16;
  const int qrow = t0 + l15;                        // this lane's query row

  const bf16_t* Qb  = Q  + (long)b * Tn * Hn;
  const bf16_t* Kb  = K  + (long)b * Tn * Hn;
  const bf16_t* Vtb = Vt + (long)b * Hn * Tn;

  // Q^T B-fragments: lane = query column, two c-chunks of 32.
  v16bf qb[2];
  {
    const bf16_t* qr = Qb + (long)qrow * Hn + koffB;
    qb[0] = load_bfrag(qr);
    qb[1] = load_bfrag(qr + 32);
  }

  const v8f z = {0.f,0.f,0.f,0.f,0.f,0.f,0.f,0.f};
  v8f o[4] = {z, z, z, z};      // O^T: 64h x 16q; reg r -> h=16*nt+r+rbase
  float m = NEGINF, lsum = 0.f; // per-lane (per-query) stats

  const int jend = t0 + 16;                         // keys 0 .. t0+15 needed
  for (int j = 0; j < jend; j += 32) {
    // ---- stage K (A-frag) and V^T (A-frag) up front ---------------------
    v16bf kfrag[4];
    #pragma unroll
    for (int nt = 0; nt < 2; ++nt) {
      const bf16_t* kr = Kb + (long)(j + 16 * nt + l15) * Hn + koffA;
      kfrag[2 * nt + 0] = load_afrag(kr);
      kfrag[2 * nt + 1] = load_afrag(kr + 32);
    }
    v16bf vfrag[4];
    #pragma unroll
    for (int nt = 0; nt < 4; ++nt)
      vfrag[nt] = load_afrag(Vtb + (long)(l15 + 16 * nt) * Tn + j + koffA);

    // ---- S^T = K Q^T : two 16-key row tiles -----------------------------
    v8f st[2] = {z, z};
    st[0] = wmma_bf16(kfrag[0], qb[0], st[0]);
    st[0] = wmma_bf16(kfrag[1], qb[1], st[0]);
    st[1] = wmma_bf16(kfrag[2], qb[0], st[1]);
    st[1] = wmma_bf16(kfrag[3], qb[1], st[1]);

    // ---- scale, causal mask, per-lane online softmax --------------------
    float rm = NEGINF;
    #pragma unroll
    for (int nt = 0; nt < 2; ++nt)
      #pragma unroll
      for (int r = 0; r < 8; ++r) {
        const int key = j + 16 * nt + r + rbase;
        float v = st[nt][r] * SCALE;
        if (key > qrow) v = NEGINF;
        st[nt][r] = v;
        rm = fmaxf(rm, v);
      }
    rm = fmaxf(rm, __shfl_xor(rm, 16, 32));         // merge key halves
    const float mn    = fmaxf(m, rm);
    const float alpha = __expf(m - mn);
    m = mn;

    float rs = 0.f;
    #pragma unroll
    for (int nt = 0; nt < 2; ++nt)
      #pragma unroll
      for (int r = 0; r < 8; ++r) {
        const float p = __expf(st[nt][r] - mn);
        st[nt][r] = p;
        rs += p;
      }
    rs += __shfl_xor(rs, 16, 32);                   // merge key halves
    lsum = lsum * alpha + rs;
    #pragma unroll
    for (int nt = 0; nt < 4; ++nt)
      #pragma unroll
      for (int r = 0; r < 8; ++r) o[nt][r] *= alpha;

    // ---- P -> LDS [query][key]: per-lane contiguous, 2x ds_store_b128 ---
    #pragma unroll
    for (int nt = 0; nt < 2; ++nt) {
      v8bf pk;
      #pragma unroll
      for (int r = 0; r < 8; ++r) pk[r] = (__bf16)st[nt][r];
      *(v8bf*)&pbuf[l15 * 32 + 16 * nt + rbase] = pk;
    }
    asm volatile("s_wait_dscnt 0" ::: "memory");    // LDS in-order per wave
    // P^T B-fragment: lane = query col, 16 contiguous keys.
    const v16bf pa = load_bfrag(&pbuf[l15 * 32 + koffB]);

    // ---- O^T += V^T P^T -------------------------------------------------
    #pragma unroll
    for (int nt = 0; nt < 4; ++nt)
      o[nt] = wmma_bf16(vfrag[nt], pa, o[nt]);
  }

  // ---- normalize and store: per-lane contiguous 8-float runs ------------
  const float inv = 1.0f / lsum;
  float* ob = out + ((long)b * Tn + qrow) * Hn;     // this query's output row
  #pragma unroll
  for (int nt = 0; nt < 4; ++nt)
    #pragma unroll
    for (int r = 0; r < 8; ++r)
      ob[16 * nt + r + rbase] = o[nt][r] * inv;
}

// ---------------------------------------------------------------------------
extern "C" void kernel_launch(void* const* d_in, const int* in_sizes, int n_in,
                              void* d_out, int out_size, void* d_ws, size_t ws_size,
                              hipStream_t stream) {
  const float* x  = (const float*)d_in[0];   // [B,T,C]
  const float* Wq = (const float*)d_in[1];   // [H,C]
  const float* Wk = (const float*)d_in[2];
  const float* Wv = (const float*)d_in[3];

  // Workspace (bf16): W q/k/v (3*64K elems), Q,K [B*T,64], Vt [B,H,T]. ~6.7 MB.
  bf16_t* Wqb = (bf16_t*)d_ws;
  bf16_t* Wkb = Wqb + (size_t)Hn * Cn;
  bf16_t* Wvb = Wkb + (size_t)Hn * Cn;
  bf16_t* Qs  = Wvb + (size_t)Hn * Cn;
  bf16_t* Ks  = Qs  + (size_t)Bn * Tn * Hn;
  bf16_t* Vts = Ks  + (size_t)Bn * Tn * Hn;

  cvt_weights_kernel<<<(Hn * Cn + 255) / 256, 256, 0, stream>>>(Wq, Wk, Wv, Wqb);

  const int tiles = Bn * Tn / 16;                   // 1024
  qkv_proj_kernel<<<tiles / 4, 128, 0, stream>>>(x, Wqb, Wkb, Wvb, Qs, Ks, Vts);

  flash_attn_kernel<<<tiles, 32, 0, stream>>>(Qs, Ks, Vts, (float*)d_out);
}
